// RecurrentLayer_54219667145480
// MI455X (gfx1250) — matrix operand order repaired and checked by reading, not compiled, not run on hardware
//
#include <hip/hip_runtime.h>

// ---------------------------------------------------------------------------
// Leaky ReLU RNN scan for MI455X (gfx1250, wave32, WMMA).
//   states[t] = relu(0.9*h + 0.1*(h@W_hh.T + b_hh + x_t@W_in.T + b_in))
//
// Phase A (parallel): pre[t,b,n] = x@W_in.T + b_in + b_hh  -> written to d_out
//   (131072 16x16 tiles, full-machine WMMA GEMM; d_out[t] is read+overwritten
//    by the SAME wave later in the scan, and 128 MB fits in the 192 MB L2)
// Phase B (serial, persistent kernel, grid barrier between 256 steps):
//   * 64 blocks x 256 thr (always co-resident); monotonic-counter barrier
//   * recurrent GEMM only: K=1024 split into 4 independent WMMA accumulator
//     chains (8 deep each) to cut per-step matrix-pipe latency ~4x
//   * block's 16 W_hh rows held in LDS as bf16 (t-invariant, shared by all
//     8 waves) -> B operands via ds_load_b128
//   * f32 integrator state in VGPRs across the whole scan (tile assignment
//     is t-invariant; WMMA C/D layout == state layout)
// Workspace (~17.1 MB): input bf16 | W_in bf16 | h ping/pong bf16 | ctr
// ---------------------------------------------------------------------------

#define T_STEPS 256
#define BATCH   128
#define IDIM    256
#define HDIM    1024
#define NBLK    64          // scan: HDIM / 16 output-column tiles
#define LPAD    8           // LDS row padding (elements) to skew banks
#define ALPHA_F 0.1f
#define OMA_F   0.9f        // 1 - alpha

typedef __attribute__((ext_vector_type(16))) __bf16 v16bf;
typedef __attribute__((ext_vector_type(8)))  float  v8f;

__device__ __forceinline__ __bf16 f2bf(float f) {
  // round-to-nearest-even f32 -> bf16
  unsigned int u = __float_as_uint(f);
  unsigned int r = (u + 0x7FFFu + ((u >> 16) & 1u)) >> 16;
  unsigned short s = (unsigned short)r;
  __bf16 b;
  __builtin_memcpy(&b, &s, sizeof(b));
  return b;
}

// Elementwise f32 -> bf16 staging.
__global__ void cvt_f32_to_bf16(const float* __restrict__ src,
                                __bf16* __restrict__ dst, int n) {
  int i = blockIdx.x * blockDim.x + threadIdx.x;
  if (i < n) dst[i] = f2bf(src[i]);
}

// Broadcast h0[H] into bf16 h-buffer [B,H] (t=0 A-operand).
__global__ void init_h_bf16(const float* __restrict__ h0,
                            __bf16* __restrict__ dst) {
  int i = blockIdx.x * blockDim.x + threadIdx.x;  // B*H threads
  dst[i] = f2bf(h0[i & (HDIM - 1)]);
}

__global__ void reset_ctr(unsigned int* ctr) { *ctr = 0u; }

// ---------------------------------------------------------------------------
// Phase A: pre = x @ W_in.T + (b_in + b_hh), over all T*B rows. One wave per
// 16x16 tile, 8 waves/block. Tiles: (T*B/16)=2048 m-tiles x 64 n-tiles.
__global__ __launch_bounds__(256)
void xproj_gemm(const __bf16* __restrict__ x_bf,    // [T*B, I] bf16
                const __bf16* __restrict__ Win_bf,  // [H, I]  bf16
                const float*  __restrict__ b_in,    // [H]
                const float*  __restrict__ b_hh,    // [H]
                float*        __restrict__ out)     // [T*B, H] f32 (= d_out)
{
  const int wave = threadIdx.x >> 5;
  const int lane = threadIdx.x & 31;
  const int half = lane >> 4;
  const int l16  = lane & 15;

  const int tile = blockIdx.x * 8 + wave;   // 0 .. 131071
  const int n0 = (tile & 63) * 16;
  const int m0 = (tile >> 6) * 16;

  v8f acc = {};
  const __bf16* arow = x_bf   + (size_t)(m0 + l16) * IDIM + half * 16;
  const __bf16* brow = Win_bf + (size_t)(n0 + l16) * IDIM + half * 16;
  for (int k0 = 0; k0 < IDIM; k0 += 32) {
    v16bf a, b;
    __builtin_memcpy(&a,             arow + k0,      32);
    __builtin_memcpy(&b,             brow + k0,      32);
    acc = __builtin_amdgcn_wmma_f32_16x16x32_bf16(
        false, a, false, b, (short)0, acc, false, false);
  }

  const int   n    = n0 + l16;
  const float bias = b_in[n] + b_hh[n];
#pragma unroll
  for (int v = 0; v < 8; ++v) {
    const int m = m0 + v + half * 8;
    out[(size_t)m * HDIM + n] = acc[v] + bias;
  }
}

// ---------------------------------------------------------------------------
// Phase B: persistent serial scan.
__global__ __launch_bounds__(256)
void rnn_scan(const float*  __restrict__ Whh,     // [H,H] f32
              const float*  __restrict__ h0,      // [H]
              float*        __restrict__ out,     // [T,B,H]: pre -> states
              __bf16*       __restrict__ hbuf0,   // [B,H] bf16 ping
              __bf16*       __restrict__ hbuf1,   // [B,H] bf16 pong
              unsigned int* __restrict__ ctr)     // grid barrier counter
{
  __shared__ __bf16 sWhh[16][HDIM + LPAD];  // this block's 16 W_hh rows, bf16

  const int wave = threadIdx.x >> 5;
  const int lane = threadIdx.x & 31;
  const int half = lane >> 4;
  const int l16  = lane & 15;

  const int n0 = blockIdx.x * 16; // output-column tile (shared by all 8 waves)
  const int m0 = wave * 16;       // batch tile (8 waves * 16 = 128 = B)

  // one-time: convert this block's W_hh rows to bf16 in LDS
  for (int i = threadIdx.x; i < 16 * HDIM; i += 256) {
    const int r = i >> 10, k = i & (HDIM - 1);
    sWhh[r][k] = f2bf(Whh[(size_t)(n0 + r) * HDIM + k]);
  }
  __syncthreads();

  const int n = n0 + l16;

  // f32 integrator state in VGPRs (C/D layout: VGPR v -> M = m0 + v + 8*half)
  float hstate[8];
#pragma unroll
  for (int v = 0; v < 8; ++v) hstate[v] = h0[n];

  const __bf16* sb = &sWhh[l16][half * 16];  // B-frag column base in LDS

  for (int t = 0; t < T_STEPS; ++t) {
    const __bf16* hin   = (t & 1) ? hbuf1 : hbuf0;
    __bf16*       hnext = (t & 1) ? hbuf0 : hbuf1;

    // ---- recurrent GEMM, 4 independent accumulator chains (K interleaved) -
    v8f acc0 = {}, acc1 = {}, acc2 = {}, acc3 = {};
    const __bf16* arow = hin + (size_t)(m0 + l16) * HDIM + half * 16;
    for (int k0 = 0; k0 < HDIM; k0 += 128) {
      v16bf a, b;
      __builtin_memcpy(&a,             arow + k0,           32);
      __builtin_memcpy(&b,             sb   + k0,           32);
      acc0 = __builtin_amdgcn_wmma_f32_16x16x32_bf16(false, a, false, b, (short)0, acc0, false, false);
      __builtin_memcpy(&a,             arow + k0 + 32,      32);
      __builtin_memcpy(&b,             sb   + k0 + 32,      32);
      acc1 = __builtin_amdgcn_wmma_f32_16x16x32_bf16(false, a, false, b, (short)0, acc1, false, false);
      __builtin_memcpy(&a,             arow + k0 + 64,      32);
      __builtin_memcpy(&b,             sb   + k0 + 64,      32);
      acc2 = __builtin_amdgcn_wmma_f32_16x16x32_bf16(false, a, false, b, (short)0, acc2, false, false);
      __builtin_memcpy(&a,             arow + k0 + 96,      32);
      __builtin_memcpy(&b,             sb   + k0 + 96,      32);
      acc3 = __builtin_amdgcn_wmma_f32_16x16x32_bf16(false, a, false, b, (short)0, acc3, false, false);
    }

    // ---- leaky integration: pre (from phase A) lives in out[t] -----------
    float* out_t = out + (size_t)t * BATCH * HDIM;
#pragma unroll
    for (int v = 0; v < 8; ++v) {
      const int m = m0 + v + half * 8;
      const float rec = (acc0[v] + acc1[v]) + (acc2[v] + acc3[v]);
      const float pre = out_t[(size_t)m * HDIM + n];  // x@W_in.T + b_in + b_hh
      float val = OMA_F * hstate[v] + ALPHA_F * (rec + pre);
      val = fmaxf(val, 0.0f);
      hstate[v] = val;
      out_t[(size_t)m * HDIM + n] = val;
      hnext[(size_t)m * HDIM + n] = f2bf(val);
    }

    // ---- grid-wide barrier: release h stores, arrive, spin, acquire ------
    __threadfence();
    __syncthreads();
    if (threadIdx.x == 0) {
      __hip_atomic_fetch_add(ctr, 1u, __ATOMIC_ACQ_REL, __HIP_MEMORY_SCOPE_AGENT);
      const unsigned int target = (unsigned int)NBLK * (unsigned int)(t + 1);
      while (__hip_atomic_load(ctr, __ATOMIC_ACQUIRE, __HIP_MEMORY_SCOPE_AGENT) < target) {
        __builtin_amdgcn_s_sleep(1);
      }
    }
    __syncthreads();
    __threadfence();
  }
}

// ---------------------------------------------------------------------------
extern "C" void kernel_launch(void* const* d_in, const int* in_sizes, int n_in,
                              void* d_out, int out_size, void* d_ws, size_t ws_size,
                              hipStream_t stream) {
  const float* input = (const float*)d_in[0];  // [T,B,I]
  const float* W_in  = (const float*)d_in[1];  // [H,I]
  const float* b_in  = (const float*)d_in[2];  // [H]
  const float* W_hh  = (const float*)d_in[3];  // [H,H]
  const float* b_hh  = (const float*)d_in[4];  // [H]
  const float* h0    = (const float*)d_in[5];  // [H]
  float* out = (float*)d_out;                  // [T,B,H]

  const size_t nX   = (size_t)T_STEPS * BATCH * IDIM;  // 8M elems
  const size_t nWin = (size_t)HDIM * IDIM;             // 256K elems
  const size_t nH   = (size_t)BATCH * HDIM;            // 128K elems

  char* ws = (char*)d_ws;
  __bf16*       x_bf   = (__bf16*)(ws);                          // 16 MB
  __bf16*       Win_bf = (__bf16*)(ws + nX * 2);                 // 512 KB
  __bf16*       hbuf0  = (__bf16*)(ws + (nX + nWin) * 2);        // 256 KB
  __bf16*       hbuf1  = hbuf0 + nH;                             // 256 KB
  unsigned int* ctr    = (unsigned int*)(ws + (nX + nWin + 2 * nH) * 2);

  // Staging (every call; ws is scratch, deterministic).
  cvt_f32_to_bf16<<<(int)((nX + 255) / 256), 256, 0, stream>>>(input, x_bf, (int)nX);
  cvt_f32_to_bf16<<<(int)((nWin + 255) / 256), 256, 0, stream>>>(W_in, Win_bf, (int)nWin);
  init_h_bf16<<<(int)(nH / 256), 256, 0, stream>>>(h0, hbuf0);
  reset_ctr<<<1, 1, 0, stream>>>(ctr);

  // Phase A: parallel input-projection GEMM into d_out (pre-activations).
  xproj_gemm<<<16384, 256, 0, stream>>>(x_bf, Win_bf, b_in, b_hh, out);

  // Phase B: 256-step serial scan in one persistent kernel.
  rnn_scan<<<NBLK, 256, 0, stream>>>(W_hh, h0, out, hbuf0, hbuf1, ctr);
}
